// LSTM_classifier_49254684950850
// MI455X (gfx1250) — compile-verified, matrix operand order checked by reading, not run on hardware
//
#include <hip/hip_runtime.h>
#include <hip/hip_bf16.h>
#include <math.h>

typedef __attribute__((ext_vector_type(16))) _Float16 v16h;
typedef __attribute__((ext_vector_type(8)))  float    v8f;

#define BATCH   65536
#define IN_DIM  64
#define H_DIM   128
#define OUT_DIM 2
#define ROWS_PER_BLOCK 128   // 8 waves * 16 rows

__device__ __forceinline__ float sigmoid_f(float x) {
  return 1.0f / (1.0f + __expf(-x));
}
__device__ __forceinline__ float tanh_f(float x) {
  float e2 = __expf(-2.0f * fabsf(x));
  float t  = (1.0f - e2) / (1.0f + e2);
  return copysignf(t, x);
}

// gb[g*128 + n] = bx_g[n] + sum_k h0[k] * Wh_g[n*H + k]   (g: 0=f,1=i,2=o,3=c)
__global__ void lstm_prep_kernel(const float* __restrict__ h0,
    const float* __restrict__ bxf, const float* __restrict__ Whf,
    const float* __restrict__ bxi, const float* __restrict__ Whi,
    const float* __restrict__ bxo, const float* __restrict__ Who,
    const float* __restrict__ bxc, const float* __restrict__ Whc,
    float* __restrict__ gb)
{
  int t = blockIdx.x * blockDim.x + threadIdx.x;
  if (t >= 4 * H_DIM) return;
  int g = t >> 7;
  int n = t & (H_DIM - 1);
  const float* Wh = (g == 0) ? Whf : (g == 1) ? Whi : (g == 2) ? Who : Whc;
  const float* bx = (g == 0) ? bxf : (g == 1) ? bxi : (g == 2) ? bxo : bxc;
  float s = bx[n];
  #pragma unroll 4
  for (int k = 0; k < H_DIM; ++k) s += h0[k] * Wh[n * H_DIM + k];
  gb[t] = s;
}

// One block = 256 threads = 8 waves; wave w handles rows [blk*128 + w*16, +16).
// Gate pre-activations via v_wmma_f32_16x16x32_f16, weights staged in LDS (f16 B-layout).
__launch_bounds__(256)
__global__ void lstm_main_kernel(const float* __restrict__ x,
                                 const float* __restrict__ c0,
                                 const float* __restrict__ Wxf,
                                 const float* __restrict__ Wxi,
                                 const float* __restrict__ Wxo,
                                 const float* __restrict__ Wxc,
                                 const float* __restrict__ Wy,
                                 const float* __restrict__ by,
                                 const float* __restrict__ gb,
                                 float* __restrict__ y)
{
  // 64 tiles of 512 f16 each: tile T = ((g*8 + ntile)*2 + kstep), lane-major inside.
  __shared__ __align__(32) _Float16 Wlds[4 * 8 * 2 * 512];   // 64 KB
  __shared__ float hlds[8][16][17];                          // per-wave h staging (pad=17)

  const int tid  = threadIdx.x;
  const int lane = tid & 31;
  const int wave = tid >> 5;
  const int half = lane >> 4;   // 0: K-group low / rows 0-7 ; 1: K-group high / rows 8-15
  const int l16  = lane & 15;

  // ---- stage Wx{f,i,o,c} -> LDS in f16 WMMA B-layout (each wave: 8 tiles) ----
  #pragma unroll
  for (int tt = 0; tt < 8; ++tt) {
    const int T  = wave * 8 + tt;
    const int g  = T >> 4;          // 16 tiles per gate
    const int nt = (T >> 1) & 7;    // column tile
    const int ks = T & 1;           // k-step
    const float* W = (g == 0) ? Wxf : (g == 1) ? Wxi : (g == 2) ? Wxo : Wxc;
    const int N     = nt * 16 + l16;        // output column (hidden unit)
    const int kbase = ks * 32 + half * 8;
    _Float16* dst = &Wlds[T * 512 + lane * 16];
    #pragma unroll
    for (int e = 0; e < 16; ++e) {
      const int K = kbase + e + ((e >= 8) ? 8 : 0);  // ISA 16-bit K striping
      dst[e] = (_Float16)W[N * IN_DIM + K];          // B(k,n) = Wx[n,k]  (x @ Wx.T)
    }
  }
  __syncthreads();

  const int rowBase = blockIdx.x * ROWS_PER_BLOCK + wave * 16;

  // ---- load A tiles (16 rows x 64 K) as two 16x32 f16 tiles in registers ----
  v16h a0, a1;
  {
    const float* xr = x + (rowBase + l16) * IN_DIM;
    const int kb = half * 8;
    #pragma unroll
    for (int e = 0; e < 16; ++e) {
      const int K = kb + e + ((e >= 8) ? 8 : 0);
      a0[e] = (_Float16)xr[K];
      a1[e] = (_Float16)xr[32 + K];
    }
  }

  float yacc = 0.0f;   // partial y for (row = l16, out = half)

  for (int nt = 0; nt < 8; ++nt) {
    // ---- 4 gate pre-activations for this 16x16 tile ----
    v8f acc[4];
    #pragma unroll
    for (int g = 0; g < 4; ++g) {
      const _Float16* bbase = &Wlds[(((g << 3) | nt) << 1) * 512 + lane * 16];
      v16h b0 = *(const v16h*)(bbase);
      v16h b1 = *(const v16h*)(bbase + 512);
      v8f c = {};
      c = __builtin_amdgcn_wmma_f32_16x16x32_f16(false, a0, false, b0, (short)0, c, false, false);
      c = __builtin_amdgcn_wmma_f32_16x16x32_f16(false, a1, false, b1, (short)0, c, false, false);
      const float s = gb[(g << 7) + (nt << 4) + l16];  // bias + h0@Wh.T, same N for all 8 rows
      #pragma unroll
      for (int v = 0; v < 8; ++v) c[v] += s;
      acc[g] = c;
    }

    // ---- c0 tile in C/D layout: lane holds column N, VGPR v holds row half*8+v ----
    const int N = nt * 16 + l16;
    v8f c0t;
    #pragma unroll
    for (int v = 0; v < 8; ++v)
      c0t[v] = c0[(rowBase + half * 8 + v) * H_DIM + N];

    // ---- elementwise LSTM cell; stage h through per-wave LDS ----
    #pragma unroll
    for (int v = 0; v < 8; ++v) {
      const float fg = sigmoid_f(acc[0][v]);
      const float ig = sigmoid_f(acc[1][v]);
      const float og = sigmoid_f(acc[2][v]);
      const float ct = tanh_f(acc[3][v]);
      const float cc = fg * c0t[v] + ig * ct;
      const float hh = og * tanh_f(cc);
      hlds[wave][half * 8 + v][l16] = hh;
    }
    // wave-synchronous LDS: HW keeps DS ops in order within a wave; fence the compiler.
    asm volatile("" ::: "memory");

    // ---- output head: y[row, j] partial, lane = (row=l16, j=half) ----
    #pragma unroll
    for (int e = 0; e < 16; ++e)
      yacc += hlds[wave][l16][e] * Wy[half * H_DIM + nt * 16 + e];
    asm volatile("" ::: "memory");
  }

  y[(rowBase + l16) * OUT_DIM + half] = yacc + by[half];
}

extern "C" void kernel_launch(void* const* d_in, const int* in_sizes, int n_in,
                              void* d_out, int out_size, void* d_ws, size_t ws_size,
                              hipStream_t stream) {
  const float* x   = (const float*)d_in[0];
  const float* h0  = (const float*)d_in[1];
  const float* c0  = (const float*)d_in[2];
  const float* Wxf = (const float*)d_in[3];
  const float* bxf = (const float*)d_in[4];
  const float* Whf = (const float*)d_in[5];
  const float* Wxi = (const float*)d_in[6];
  const float* bxi = (const float*)d_in[7];
  const float* Whi = (const float*)d_in[8];
  const float* Wxo = (const float*)d_in[9];
  const float* bxo = (const float*)d_in[10];
  const float* Who = (const float*)d_in[11];
  const float* Wxc = (const float*)d_in[12];
  const float* bxc = (const float*)d_in[13];
  const float* Whc = (const float*)d_in[14];
  const float* Wy  = (const float*)d_in[15];
  const float* by  = (const float*)d_in[16];
  float* yout = (float*)d_out;
  float* gb   = (float*)d_ws;   // 512 floats

  lstm_prep_kernel<<<1, 512, 0, stream>>>(h0, bxf, Whf, bxi, Whi, bxo, Who, bxc, Whc, gb);
  lstm_main_kernel<<<BATCH / ROWS_PER_BLOCK, 256, 0, stream>>>(
      x, c0, Wxf, Wxi, Wxo, Wxc, Wy, by, gb, yout);
}
